// UDFAGNNlayer_79577154060384
// MI455X (gfx1250) — matrix operand order repaired and checked by reading, not compiled, not run on hardware
//
#include <hip/hip_runtime.h>
#include <hip/hip_bf16.h>

typedef __attribute__((ext_vector_type(16))) _Float16 v16h;
typedef __attribute__((ext_vector_type(8)))  _Float16 v8h;
typedef __attribute__((ext_vector_type(4)))  _Float16 v4h;
typedef __attribute__((ext_vector_type(8)))  float    v8f;

#define F 128

// ---------------------------------------------------------------------------
// Kernel 1: L2-normalize rows of x, store as f16 (halves gather traffic).
// One wave (32 lanes) per row; lane handles 4 consecutive floats.
// ---------------------------------------------------------------------------
__global__ void agnn_normalize_f16(const float* __restrict__ x,
                                   _Float16* __restrict__ xh, int N) {
  const int wave = threadIdx.x >> 5;
  const int lane = threadIdx.x & 31;
  const long row = (long)blockIdx.x * 8 + wave;
  if (row >= N) return;

  const float4 v = ((const float4*)(x + row * F))[lane];
  float ss = v.x * v.x + v.y * v.y + v.z * v.z + v.w * v.w;
#pragma unroll
  for (int off = 16; off > 0; off >>= 1) ss += __shfl_xor(ss, off, 32);
  const float r = rsqrtf(fmaxf(ss, 1e-12f));

  v4h h;
  h[0] = (_Float16)(v.x * r);
  h[1] = (_Float16)(v.y * r);
  h[2] = (_Float16)(v.z * r);
  h[3] = (_Float16)(v.w * r);
  *(v4h*)(xh + row * F + lane * 4) = h;
}

// ---------------------------------------------------------------------------
// Kernel 2: per-edge similarity via WMMA. One wave handles 16 edges:
//   D = A(16x128 f16, rows = xn[row_e]) x B(128x16 f16, cols = xn[col_e])
// as 4 chained v_wmma_f32_16x16x32_f16; diag(D) = the 16 edge dots.
// Writes ex[e] = exp(beta * dot) to workspace.
// ---------------------------------------------------------------------------
__global__ void agnn_edge_scores_wmma(const _Float16* __restrict__ xh,
                                      const int* __restrict__ row_idx,
                                      const int* __restrict__ col_idx,
                                      const float* __restrict__ beta,
                                      float* __restrict__ ex,
                                      int E, int ntiles) {
  __shared__ float sD[8][16][16];

  const int wave = threadIdx.x >> 5;
  const int lane = threadIdx.x & 31;
  const long tile = (long)blockIdx.x * 8 + wave;
  const bool active = tile < ntiles;

  const int  m    = lane & 15;      // edge-in-tile this lane serves
  const int  hi   = lane >> 4;      // which K half-group this lane holds
  const long eb   = tile * 16;

  // prefetch upcoming index stream into L2 (speculative, drop-on-fault)
  __builtin_prefetch(row_idx + eb + 1024, 0, 1);
  __builtin_prefetch(col_idx + eb + 1024, 0, 1);

  v8f c = {};
  if (active) {
    long ea = eb + m;
    if (ea >= E) ea = E - 1;                 // clamp tail gathers (safe)
    const long rnode = row_idx[ea];
    const long cnode = col_idx[ea];
    const _Float16* arow = xh + rnode * F;
    const _Float16* brow = xh + cnode * F;

#pragma unroll
    for (int s = 0; s < 4; ++s) {
      const int kb = s * 32;
      // A fragment (16x32, contiguous layout):
      //  lane<16 : halves0-15 = K kb+0..15
      //  lane>=16: halves0-15 = K kb+16..31
      v16h a = *(const v16h*)(arow + kb + hi * 16);

      // B fragment (32x16, contiguous layout):
      //  lane<16 : K kb+0..15 of column N=lane ; lane>=16: K kb+16..31
      v16h b = *(const v16h*)(brow + kb + hi * 16);

      c = __builtin_amdgcn_wmma_f32_16x16x32_f16(
          /*neg_a=*/false, a, /*neg_b=*/false, b,
          /*c_mod=*/(short)0, c, /*reuse_a=*/false, /*reuse_b=*/false);
    }

    // scatter accumulator to LDS: VGPR r of lane l holds D[M][N] with
    //  M = (l<16 ? r : r+8), N = l&15
#pragma unroll
    for (int r = 0; r < 8; ++r) {
      const int M = (lane < 16) ? r : (r + 8);
      sD[wave][M][m] = c[r];
    }
  }
  __syncthreads();

  if (active && lane < 16) {
    const long e = eb + lane;
    if (e < E) {
      const float dot = sD[wave][lane][lane];   // diagonal element
      ex[e] = expf(beta[0] * dot);
    }
  }
}

// ---------------------------------------------------------------------------
// Kernel 3: per-destination-row softmax-normalize + aggregate.
// One block (128 threads = 1 feature each) per row; edge range found by
// binary search in the sorted row_idx. No atomics -> deterministic.
// ---------------------------------------------------------------------------
__device__ __forceinline__ int lower_bound_i32(const int* __restrict__ a,
                                               int n, int key) {
  int lo = 0, hi = n;
  while (lo < hi) {
    int mid = (lo + hi) >> 1;
    if (a[mid] < key) lo = mid + 1; else hi = mid;
  }
  return lo;
}

__global__ void agnn_aggregate(const float* __restrict__ x,
                               const int* __restrict__ row_idx,
                               const int* __restrict__ col_idx,
                               const float* __restrict__ ex,
                               float* __restrict__ out, int E) {
  const int i = blockIdx.x;
  const int f = threadIdx.x;   // 0..127

  const int lo = lower_bound_i32(row_idx, E, i);
  const int hi = lower_bound_i32(row_idx, E, i + 1);

  float acc = 0.0f, ssum = 0.0f;
  for (int e = lo; e < hi; ++e) {
    const float w = ex[e];                       // broadcast across lanes
    ssum += w;
    acc  += w * x[(long)col_idx[e] * F + f];     // coalesced 512B row read
  }
  out[(long)i * F + f] = (hi > lo) ? (acc / ssum) : 0.0f;
}

// ---------------------------------------------------------------------------
extern "C" void kernel_launch(void* const* d_in, const int* in_sizes, int n_in,
                              void* d_out, int out_size, void* d_ws, size_t ws_size,
                              hipStream_t stream) {
  const float* x     = (const float*)d_in[0];
  const float* beta  = (const float*)d_in[1];
  const int*   rowi  = (const int*)  d_in[2];
  const int*   coli  = (const int*)  d_in[3];
  float*       out   = (float*)d_out;

  const int N = in_sizes[0] / F;
  const int E = in_sizes[2];

  // workspace layout: [ xh: N*F f16 ][ ex: E f32 ]
  _Float16* xh = (_Float16*)d_ws;
  size_t xh_bytes = ((size_t)N * F * sizeof(_Float16) + 255) & ~(size_t)255;
  float* ex = (float*)((char*)d_ws + xh_bytes);

  // 1) normalize -> f16
  {
    int blocks = (N + 7) / 8;   // 8 waves (rows) per 256-thread block
    agnn_normalize_f16<<<blocks, 256, 0, stream>>>(x, xh, N);
  }
  // 2) edge scores via WMMA (16 edges per wave)
  {
    int ntiles = (E + 15) / 16;
    int blocks = (ntiles + 7) / 8;
    agnn_edge_scores_wmma<<<blocks, 256, 0, stream>>>(xh, rowi, coli, beta,
                                                      ex, E, ntiles);
  }
  // 3) per-row normalize + aggregate
  {
    agnn_aggregate<<<N, F, 0, stream>>>(x, rowi, coli, ex, out, E);
  }
}
